// MGAN_22789096473273
// MI455X (gfx1250) — matrix operand order repaired and checked by reading, not compiled, not run on hardware
//
#include <hip/hip_runtime.h>
#include <stdint.h>
#include <math.h>

// ---------------- problem constants ----------------
#define Bc   256
#define Sc   128
#define Ac   8
#define Lc   32
#define Dc   300
#define Hc   300
#define H2c  600
#define DPc  320          // padded K for input projection (300 -> 320)
#define HPc  320          // padded K for hidden projection
#define JPc  320          // padded per-gate output width
#define NW2  (4*JPc)      // padded weight rows (i,f,g,o blocks of 320)
#define NCc  3
#define DQ   328          // LDS row pitch (elements): 328*2B=656B=41 banks -> conflict-free
#define USE_ASYNC_LDS 1   // gfx1250 GLOBAL_LOAD_ASYNC_TO_LDS path

typedef __bf16 bf16;
typedef bf16  v16bf __attribute__((ext_vector_type(16)));
typedef float v8f   __attribute__((ext_vector_type(8)));
typedef unsigned int u32x4 __attribute__((ext_vector_type(4)));

struct B32x2 { u32x4 lo, hi; };

__device__ __forceinline__ v16bf pack_frag(const bf16* p0, const bf16* p1) {
  B32x2 f;
  f.lo = *(const u32x4*)p0;
  f.hi = *(const u32x4*)p1;
  return __builtin_bit_cast(v16bf, f);
}
// A fragment (16x32 bf16): lane = (half, m). VGPR0-3 hold K=[half*8, half*8+8),
// VGPR4-7 hold K=[16+half*8, 16+half*8+8). Row pointer is per-lane row m.
__device__ __forceinline__ v16bf load_a(const bf16* row, int k, int half) {
  const bf16* p = row + k + half * 8;
  return pack_frag(p, p + 16);
}
// B fragment (32x16 bf16): lane = (half, n); VGPR0-3 hold K=[half*8, half*8+8),
// VGPR4-7 hold K=[16+half*8, 16+half*8+8) of output column n (= staged weight row n). lrow points at the LDS row.
__device__ __forceinline__ v16bf load_b_lds(const bf16* lrow, int k, int half) {
  const bf16* p = lrow + k + half * 8;
  return pack_frag(p, p + 16);
}

__device__ __forceinline__ float sigf(float x) { return 1.f / (1.f + expf(-x)); }

// Stage 64 weight rows (4 gates x 16 j-columns of this block) x 320 elements
// into LDS with row pitch DQ. 2560 16-byte chunks over 256 threads = 10 each.
__device__ __forceinline__ void stage_weights(const bf16* __restrict__ W, int j0,
                                              int tid, bf16* smem) {
  #pragma unroll
  for (int i = 0; i < 10; i++) {
    int chunk = tid + i * 256;          // 0..2559
    int r = chunk / 40;                 // 40 chunks per 320-element row
    int col = (chunk % 40) * 8;
    int g = r >> 4, jr = r & 15;
    const bf16* gp = W + (size_t)(g * JPc + j0 + jr) * DPc + col;
    bf16* lp = smem + r * DQ + col;
#if USE_ASYNC_LDS
    unsigned lds_off = (unsigned)(uintptr_t)lp;   // low 32 bits = LDS byte offset
    asm volatile("global_load_async_to_lds_b128 %0, %1, off"
                 :: "v"(lds_off), "v"(gp) : "memory");
#else
    *(u32x4*)lp = *(const u32x4*)gp;
#endif
  }
#if USE_ASYNC_LDS
  asm volatile("s_wait_asynccnt 0x0" ::: "memory");
#endif
}

// ---------------- utility kernels ----------------
__global__ void zero32(uint32_t* p, size_t n) {
  size_t i = (size_t)blockIdx.x * blockDim.x + threadIdx.x;
  if (i < n) p[i] = 0u;
}

__global__ void seq_len_k(const int* __restrict__ tok, int* __restrict__ len, int T) {
  int b = blockIdx.x * blockDim.x + threadIdx.x;
  if (b >= Bc) return;
  int c = 0;
  for (int t = 0; t < T; t++) c += (tok[(size_t)b * T + t] != 0) ? 1 : 0;
  len[b] = c;
}

__global__ void idx_bwd_k(const int* __restrict__ len, int* __restrict__ idx, int T) {
  int i = blockIdx.x * blockDim.x + threadIdx.x;
  if (i >= Bc * T) return;
  int b = i / T, t = i % T;
  int v = len[b] - 1 - t;
  v = v < 0 ? 0 : v;
  v = v > (T - 1) ? (T - 1) : v;
  idx[i] = v;
}

__global__ void embed_k(const int* __restrict__ tok, const float* __restrict__ emb,
                        bf16* __restrict__ dst, long n) {
  long i = (long)blockIdx.x * blockDim.x + threadIdx.x;
  if (i >= n) return;
  int d = (int)(i % DPc);
  long r = i / DPc;
  float v = (d < Dc) ? emb[(size_t)tok[r] * Dc + d] : 0.f;
  dst[i] = (bf16)v;
}

// Repack (4H=1200, 300) f32 weights into gate-padded (1280, 320) bf16
__global__ void conv_w_k(const float* __restrict__ src, bf16* __restrict__ dst) {
  long i = (long)blockIdx.x * blockDim.x + threadIdx.x;
  const long n = (long)NW2 * DPc;
  if (i >= n) return;
  int k = (int)(i % DPc);
  int rr = (int)(i / DPc);
  int g = rr / JPc, j = rr % JPc;
  float v = (j < Hc && k < Dc) ? src[(size_t)(g * Hc + j) * Dc + k] : 0.f;
  dst[i] = (bf16)v;
}

// ---------------- fused LSTM step: async-staged WMMA GEMM + pointwise ----------------
// gates = x_t @ Wih^T + h @ Whh^T + bih + bhh ; then i/f/g/o nonlinearity, state
// update, output scatter. All 8 waves of a block share the same 16 hidden cols
// (j0), so gate weights are staged once into LDS via async global->LDS loads.
__global__ __launch_bounds__(256) void lstm_step(
    const bf16* __restrict__ X, const int* __restrict__ idx,
    const bf16* __restrict__ Wih2, const bf16* __restrict__ Whh2,
    const float* __restrict__ bih, const float* __restrict__ bhh,
    const bf16* __restrict__ hin, bf16* __restrict__ hout,
    float* __restrict__ hst, float* __restrict__ cst,
    float* __restrict__ outp, const int* __restrict__ len,
    int t, int Sx, int dir)
{
  __shared__ bf16 smem[64 * DQ];   // 41,984 B
  const int tid  = threadIdx.x;
  const int lane = tid & 31;
  const int wave = tid >> 5;
  const int half = lane >> 4;
  const int q16  = lane & 15;
  const int m0 = blockIdx.x * 128 + wave * 16;
  const int j0 = blockIdx.y * 16;

  // stage input-projection weights (4 gates x 16 rows) for this block
  stage_weights(Wih2, j0, tid, smem);

  v8f acc_i = {0.f,0.f,0.f,0.f,0.f,0.f,0.f,0.f};
  v8f acc_f = acc_i, acc_g = acc_i, acc_o = acc_i;

  const int row = m0 + q16;
  const int srct = idx ? idx[(size_t)row * Sx + t] : t;
  const bf16* xrow = X + ((size_t)row * Sx + srct) * DPc;
  const bf16* hrow = hin + (size_t)row * HPc;

  const bf16* l0 = smem + (0 * 16 + q16) * DQ;
  const bf16* l1 = smem + (1 * 16 + q16) * DQ;
  const bf16* l2 = smem + (2 * 16 + q16) * DQ;
  const bf16* l3 = smem + (3 * 16 + q16) * DQ;

  __syncthreads();   // staged Wih visible to all waves

  #pragma unroll
  for (int k = 0; k < DPc; k += 32) {
    v16bf a = load_a(xrow, k, half);
    acc_i = __builtin_amdgcn_wmma_f32_16x16x32_bf16(false, a, false, load_b_lds(l0, k, half), (short)0, acc_i, false, false);
    acc_f = __builtin_amdgcn_wmma_f32_16x16x32_bf16(false, a, false, load_b_lds(l1, k, half), (short)0, acc_f, false, false);
    acc_g = __builtin_amdgcn_wmma_f32_16x16x32_bf16(false, a, false, load_b_lds(l2, k, half), (short)0, acc_g, false, false);
    acc_o = __builtin_amdgcn_wmma_f32_16x16x32_bf16(false, a, false, load_b_lds(l3, k, half), (short)0, acc_o, false, false);
  }

  __syncthreads();   // all waves done reading Wih stage
  stage_weights(Whh2, j0, tid, smem);
  __syncthreads();   // staged Whh visible to all waves

  #pragma unroll
  for (int k = 0; k < HPc; k += 32) {
    v16bf a = load_a(hrow, k, half);
    acc_i = __builtin_amdgcn_wmma_f32_16x16x32_bf16(false, a, false, load_b_lds(l0, k, half), (short)0, acc_i, false, false);
    acc_f = __builtin_amdgcn_wmma_f32_16x16x32_bf16(false, a, false, load_b_lds(l1, k, half), (short)0, acc_f, false, false);
    acc_g = __builtin_amdgcn_wmma_f32_16x16x32_bf16(false, a, false, load_b_lds(l2, k, half), (short)0, acc_g, false, false);
    acc_o = __builtin_amdgcn_wmma_f32_16x16x32_bf16(false, a, false, load_b_lds(l3, k, half), (short)0, acc_o, false, false);
  }

  // C layout: VGPR r, lane(half,n) -> M = m0 + 8*half + r, hidden col j = j0 + n
  const int j = j0 + q16;
  const bool jok = (j < Hc);
  float bi = 0.f, bff = 0.f, bg = 0.f, bo = 0.f;
  if (jok) {
    bi  = bih[j]          + bhh[j];
    bff = bih[Hc + j]     + bhh[Hc + j];
    bg  = bih[2 * Hc + j] + bhh[2 * Hc + j];
    bo  = bih[3 * Hc + j] + bhh[3 * Hc + j];
  }
  #pragma unroll
  for (int r = 0; r < 8; r++) {
    int rw = m0 + half * 8 + r;
    float gi = sigf(acc_i[r] + bi);
    float gf = sigf(acc_f[r] + bff);
    float gg = tanhf(acc_g[r] + bg);
    float go = sigf(acc_o[r] + bo);
    if (jok) {
      int L = len[rw];
      bool msk = (t < L);
      size_t si = (size_t)rw * Hc + j;
      float cold = cst[si];
      float hold = hst[si];
      float cn = gf * cold + gi * gg;
      float hn = go * tanhf(cn);
      float h2 = msk ? hn : hold;
      float c2 = msk ? cn : cold;
      cst[si] = c2;
      hst[si] = h2;
      hout[(size_t)rw * HPc + j] = (bf16)h2;
      if (dir == 0) {
        outp[((size_t)rw * Sx + t) * H2c + j] = msk ? h2 : 0.f;
      } else if (msk) {
        outp[((size_t)rw * Sx + (L - 1 - t)) * H2c + Hc + j] = h2;
      }
    }
  }
}

// ---------------- attention / pooling kernels (f32 VALU; tiny FLOPs) ----------------
__global__ void pos_weight_k(float* __restrict__ ctx, const int* __restrict__ clen,
                             const int* __restrict__ llen, const int* __restrict__ alen) {
  size_t i = (size_t)blockIdx.x * blockDim.x + threadIdx.x;
  if (i >= (size_t)Bc * Sc * H2c) return;
  int s = (int)((i / H2c) % Sc);
  int b = (int)(i / ((size_t)H2c * Sc));
  float t = (float)s;
  float cl = (float)clen[b], ll = (float)llen[b], al = (float)alen[b];
  float denom = cl - al + 1.f;
  float w;
  if (t < ll)            w = 1.f - (ll - t) / denom;
  else if (t < ll + al)  w = 0.f;
  else if (t < cl)       w = 1.f - (t - ll - al + 1.f) / denom;
  else                   w = 0.f;
  ctx[i] *= w;
}

__global__ void avg_seq_k(const float* __restrict__ kv, const int* __restrict__ len,
                          float* __restrict__ outv, int T) {
  int i = blockIdx.x * blockDim.x + threadIdx.x;
  if (i >= Bc * H2c) return;
  int b = i / H2c, d = i % H2c;
  float acc = 0.f;
  for (int t = 0; t < T; t++) acc += kv[((size_t)b * T + t) * H2c + d];
  outv[i] = acc / (float)len[b];
}

__global__ void matvec_k(const float* __restrict__ inv, const float* __restrict__ w,
                         float* __restrict__ outv) {
  int i = blockIdx.x * blockDim.x + threadIdx.x;
  if (i >= Bc * H2c) return;
  int b = i / H2c, j = i % H2c;
  float acc = 0.f;
  for (int d = 0; d < H2c; d++) acc += inv[(size_t)b * H2c + d] * w[(size_t)d * H2c + j];
  outv[i] = acc;
}

// out[row] = sum_d kv[row,d] * (q ? q[b,d] : wv[d]),  row = b*T+s
__global__ void rowdot_k(const float* __restrict__ kv, const float* __restrict__ q,
                         const float* __restrict__ wv, float* __restrict__ outv, int T) {
  int row = blockIdx.x;
  int tid = threadIdx.x;
  int b = row / T;
  const float* r = kv + (size_t)row * H2c;
  const float* p = q ? (q + (size_t)b * H2c) : wv;
  float acc = 0.f;
  for (int d = tid; d < H2c; d += 64) acc += r[d] * p[d];
  __shared__ float red[64];
  red[tid] = acc; __syncthreads();
  for (int off = 32; off > 0; off >>= 1) {
    if (tid < off) red[tid] += red[tid + off];
    __syncthreads();
  }
  if (tid == 0) outv[row] = red[0];
}

__global__ void softmax_rows_k(float* __restrict__ x, int n) {
  int b = blockIdx.x;
  int tid = threadIdx.x;
  __shared__ float red[128];
  float v = (tid < n) ? x[(size_t)b * n + tid] : -1e30f;
  red[tid] = v; __syncthreads();
  for (int off = 64; off > 0; off >>= 1) {
    if (tid < off) red[tid] = fmaxf(red[tid], red[tid + off]);
    __syncthreads();
  }
  float mx = red[0]; __syncthreads();
  float e = (tid < n) ? expf(v - mx) : 0.f;
  red[tid] = e; __syncthreads();
  for (int off = 64; off > 0; off >>= 1) {
    if (tid < off) red[tid] += red[tid + off];
    __syncthreads();
  }
  float s = red[0];
  if (tid < n) x[(size_t)b * n + tid] = e / s;
}

__global__ void wsum_k(const float* __restrict__ alpha, const float* __restrict__ kv,
                       float* __restrict__ outv, int T) {
  int i = blockIdx.x * blockDim.x + threadIdx.x;
  if (i >= Bc * H2c) return;
  int b = i / H2c, d = i % H2c;
  float acc = 0.f;
  for (int t = 0; t < T; t++)
    acc += alpha[(size_t)b * T + t] * kv[((size_t)b * T + t) * H2c + d];
  outv[i] = acc;
}

__global__ void u_kernel(const float* __restrict__ ctx, const float* __restrict__ asp,
                         const float* __restrict__ uc, const float* __restrict__ ua,
                         const float* __restrict__ wm, const float* __restrict__ fc1b,
                         float* __restrict__ u) {
  int row = blockIdx.x;          // b*Sc + s
  int b = row / Sc;
  int tid = threadIdx.x;         // 64 threads
  const float* crow = ctx + (size_t)row * H2c;
  float p[Ac];
  #pragma unroll
  for (int a = 0; a < Ac; a++) p[a] = 0.f;
  for (int d = tid; d < H2c; d += 64) {
    float cv = crow[d] * wm[d];
    #pragma unroll
    for (int a = 0; a < Ac; a++) p[a] += cv * asp[((size_t)b * Ac + a) * H2c + d];
  }
  __shared__ float red[64][Ac];
  #pragma unroll
  for (int a = 0; a < Ac; a++) red[tid][a] = p[a];
  __syncthreads();
  for (int off = 32; off > 0; off >>= 1) {
    if (tid < off)
      for (int a = 0; a < Ac; a++) red[tid][a] += red[tid + off][a];
    __syncthreads();
  }
  if (tid < Ac)
    u[(size_t)row * Ac + tid] = red[0][tid] + uc[row] + ua[(size_t)b * Ac + tid] + fc1b[0];
}

__global__ void umax_k(const float* __restrict__ u, float* __restrict__ um) {
  int i = blockIdx.x * blockDim.x + threadIdx.x;
  if (i >= Bc * Sc) return;
  const float* p = u + (size_t)i * Ac;
  float m = p[0];
  #pragma unroll
  for (int a = 1; a < Ac; a++) m = fmaxf(m, p[a]);
  um[i] = m;
}

__global__ void softmax8_k(float* __restrict__ u) {
  int i = blockIdx.x * blockDim.x + threadIdx.x;
  if (i >= Bc * Sc) return;
  float* p = u + (size_t)i * Ac;
  float m = p[0];
  #pragma unroll
  for (int a = 1; a < Ac; a++) m = fmaxf(m, p[a]);
  float e[Ac], s = 0.f;
  #pragma unroll
  for (int a = 0; a < Ac; a++) { e[a] = expf(p[a] - m); s += e[a]; }
  #pragma unroll
  for (int a = 0; a < Ac; a++) p[a] = e[a] / s;
}

__global__ void mfc_kernel(const float* __restrict__ u, const float* __restrict__ asp,
                           float* __restrict__ outv) {
  int i = blockIdx.x * blockDim.x + threadIdx.x;
  if (i >= Bc * H2c) return;
  int b = i / H2c, d = i % H2c;
  float av[Ac];
  #pragma unroll
  for (int a = 0; a < Ac; a++) av[a] = asp[((size_t)b * Ac + a) * H2c + d];
  float acc = 0.f;
  for (int s = 0; s < Sc; s++) {
    const float* ur = u + ((size_t)b * Sc + s) * Ac;
    #pragma unroll
    for (int a = 0; a < Ac; a++) acc += ur[a] * av[a];
  }
  outv[i] = acc / (float)Sc;
}

__global__ void final_fc(const float* __restrict__ mca, const float* __restrict__ mcc,
                         const float* __restrict__ mfa, const float* __restrict__ mfc,
                         const float* __restrict__ w, const float* __restrict__ bias,
                         float* __restrict__ outv) {
  int b = blockIdx.x * blockDim.x + threadIdx.x;
  if (b >= Bc) return;
  float lg[NCc];
  for (int c = 0; c < NCc; c++) {
    const float* wc = w + (size_t)c * 4 * H2c;
    float s = bias[c];
    for (int k = 0; k < H2c; k++) {
      s += mca[(size_t)b * H2c + k] * wc[k]
         + mcc[(size_t)b * H2c + k] * wc[H2c + k]
         + mfa[(size_t)b * H2c + k] * wc[2 * H2c + k]
         + mfc[(size_t)b * H2c + k] * wc[3 * H2c + k];
    }
    lg[c] = s;
  }
  float mx = fmaxf(lg[0], fmaxf(lg[1], lg[2]));
  float e0 = expf(lg[0] - mx), e1 = expf(lg[1] - mx), e2 = expf(lg[2] - mx);
  float sm = e0 + e1 + e2;
  outv[(size_t)b * NCc + 0] = e0 / sm;
  outv[(size_t)b * NCc + 1] = e1 / sm;
  outv[(size_t)b * NCc + 2] = e2 / sm;
}

// ---------------- host driver ----------------
static inline unsigned cdivu(long a, long b) { return (unsigned)((a + b - 1) / b); }

extern "C" void kernel_launch(void* const* d_in, const int* in_sizes, int n_in,
                              void* d_out, int out_size, void* d_ws, size_t ws_size,
                              hipStream_t stream) {
  (void)in_sizes; (void)n_in; (void)out_size; (void)ws_size;

  const int*   text   = (const int*)  d_in[0];
  const int*   aspect = (const int*)  d_in[1];
  const int*   left   = (const int*)  d_in[2];
  const float* emb    = (const float*)d_in[3];
  const float* Wih_f  = (const float*)d_in[4];
  const float* Whh_f  = (const float*)d_in[5];
  const float* bih_f  = (const float*)d_in[6];
  const float* bhh_f  = (const float*)d_in[7];
  const float* Wih_b  = (const float*)d_in[8];
  const float* Whh_b  = (const float*)d_in[9];
  const float* bih_b  = (const float*)d_in[10];
  const float* bhh_b  = (const float*)d_in[11];
  const float* w1     = (const float*)d_in[12];
  const float* w2     = (const float*)d_in[13];
  const float* fc1_w  = (const float*)d_in[14];
  const float* fc1_b  = (const float*)d_in[15];
  const float* fc2_w  = (const float*)d_in[16];
  const float* fc2_b  = (const float*)d_in[17];
  float* outp = (float*)d_out;

  char* base = (char*)d_ws;
  size_t off = 0;
  auto alloc = [&](size_t bytes) -> char* {
    char* p = base + off;
    off += (bytes + 255) & ~(size_t)255;
    return p;
  };

  bf16* ctxbf   = (bf16*)alloc((size_t)Bc * Sc * DPc * 2);
  bf16* aspbf   = (bf16*)alloc((size_t)Bc * Ac * DPc * 2);
  bf16* wihf2   = (bf16*)alloc((size_t)NW2 * DPc * 2);
  bf16* whhf2   = (bf16*)alloc((size_t)NW2 * HPc * 2);
  bf16* wihb2   = (bf16*)alloc((size_t)NW2 * DPc * 2);
  bf16* whhb2   = (bf16*)alloc((size_t)NW2 * HPc * 2);
  float* hf     = (float*)alloc((size_t)Bc * Hc * 4);
  float* cf     = (float*)alloc((size_t)Bc * Hc * 4);
  float* hbst   = (float*)alloc((size_t)Bc * Hc * 4);
  float* cbst   = (float*)alloc((size_t)Bc * Hc * 4);
  bf16* hbf_f0  = (bf16*)alloc((size_t)Bc * HPc * 2);
  bf16* hbf_f1  = (bf16*)alloc((size_t)Bc * HPc * 2);
  bf16* hbf_b0  = (bf16*)alloc((size_t)Bc * HPc * 2);
  bf16* hbf_b1  = (bf16*)alloc((size_t)Bc * HPc * 2);
  float* ctx_out = (float*)alloc((size_t)Bc * Sc * H2c * 4);
  float* asp_out = (float*)alloc((size_t)Bc * Ac * H2c * 4);
  int* clen = (int*)alloc(Bc * 4);
  int* alen = (int*)alloc(Bc * 4);
  int* llen = (int*)alloc(Bc * 4);
  int* idxc = (int*)alloc((size_t)Bc * Sc * 4);
  int* idxa = (int*)alloc((size_t)Bc * Ac * 4);
  float* aavg = (float*)alloc((size_t)Bc * H2c * 4);
  float* cavg = (float*)alloc((size_t)Bc * H2c * 4);
  float* s1v  = (float*)alloc((size_t)Bc * H2c * 4);
  float* s2v  = (float*)alloc((size_t)Bc * H2c * 4);
  float* mca  = (float*)alloc((size_t)Bc * H2c * 4);
  float* mcc  = (float*)alloc((size_t)Bc * H2c * 4);
  float* mfa  = (float*)alloc((size_t)Bc * H2c * 4);
  float* mfcv = (float*)alloc((size_t)Bc * H2c * 4);
  float* sc1  = (float*)alloc((size_t)Bc * Sc * 4);
  float* sc2  = (float*)alloc((size_t)Bc * Ac * 4);
  float* ucb  = (float*)alloc((size_t)Bc * Sc * 4);
  float* uab  = (float*)alloc((size_t)Bc * Ac * 4);
  float* ub   = (float*)alloc((size_t)Bc * Sc * Ac * 4);
  float* umx  = (float*)alloc((size_t)Bc * Sc * 4);

  auto zero = [&](void* p, size_t bytes) {
    size_t n = bytes >> 2;
    zero32<<<dim3(cdivu((long)n, 256)), 256, 0, stream>>>((uint32_t*)p, n);
  };
  auto zero_states = [&]() {
    zero(hf,   (size_t)Bc * Hc * 4);
    zero(cf,   (size_t)Bc * Hc * 4);
    zero(hbst, (size_t)Bc * Hc * 4);
    zero(cbst, (size_t)Bc * Hc * 4);
    zero(hbf_f0, (size_t)Bc * HPc * 2);
    zero(hbf_f1, (size_t)Bc * HPc * 2);
    zero(hbf_b0, (size_t)Bc * HPc * 2);
    zero(hbf_b1, (size_t)Bc * HPc * 2);
  };

  // lengths + backward index maps
  seq_len_k<<<1, 256, 0, stream>>>(text, clen, Sc);
  seq_len_k<<<1, 256, 0, stream>>>(aspect, alen, Ac);
  seq_len_k<<<1, 256, 0, stream>>>(left, llen, Lc);
  idx_bwd_k<<<dim3(cdivu((long)Bc * Sc, 256)), 256, 0, stream>>>(clen, idxc, Sc);
  idx_bwd_k<<<dim3(cdivu((long)Bc * Ac, 256)), 256, 0, stream>>>(alen, idxa, Ac);

  // embedding gather -> padded bf16
  embed_k<<<dim3(cdivu((long)Bc * Sc * DPc, 256)), 256, 0, stream>>>(text, emb, ctxbf, (long)Bc * Sc * DPc);
  embed_k<<<dim3(cdivu((long)Bc * Ac * DPc, 256)), 256, 0, stream>>>(aspect, emb, aspbf, (long)Bc * Ac * DPc);

  // gate-padded bf16 weights
  conv_w_k<<<dim3(cdivu((long)NW2 * DPc, 256)), 256, 0, stream>>>(Wih_f, wihf2);
  conv_w_k<<<dim3(cdivu((long)NW2 * HPc, 256)), 256, 0, stream>>>(Whh_f, whhf2);
  conv_w_k<<<dim3(cdivu((long)NW2 * DPc, 256)), 256, 0, stream>>>(Wih_b, wihb2);
  conv_w_k<<<dim3(cdivu((long)NW2 * HPc, 256)), 256, 0, stream>>>(Whh_b, whhb2);

  zero(ctx_out, (size_t)Bc * Sc * H2c * 4);
  zero(asp_out, (size_t)Bc * Ac * H2c * 4);

  const dim3 gGemm(2, JPc / 16);  // 2 x 20 blocks of 256 threads

  // context BiLSTM
  zero_states();
  for (int t = 0; t < Sc; t++) {
    const bf16* hinF = (t & 1) ? hbf_f1 : hbf_f0;
    bf16*       houF = (t & 1) ? hbf_f0 : hbf_f1;
    lstm_step<<<gGemm, 256, 0, stream>>>(ctxbf, nullptr, wihf2, whhf2, bih_f, bhh_f,
                                         hinF, houF, hf, cf, ctx_out, clen, t, Sc, 0);
    const bf16* hinB = (t & 1) ? hbf_b1 : hbf_b0;
    bf16*       houB = (t & 1) ? hbf_b0 : hbf_b1;
    lstm_step<<<gGemm, 256, 0, stream>>>(ctxbf, idxc, wihb2, whhb2, bih_b, bhh_b,
                                         hinB, houB, hbst, cbst, ctx_out, clen, t, Sc, 1);
  }

  // aspect BiLSTM (same weights)
  zero_states();
  for (int t = 0; t < Ac; t++) {
    const bf16* hinF = (t & 1) ? hbf_f1 : hbf_f0;
    bf16*       houF = (t & 1) ? hbf_f0 : hbf_f1;
    lstm_step<<<gGemm, 256, 0, stream>>>(aspbf, nullptr, wihf2, whhf2, bih_f, bhh_f,
                                         hinF, houF, hf, cf, asp_out, alen, t, Ac, 0);
    const bf16* hinB = (t & 1) ? hbf_b1 : hbf_b0;
    bf16*       houB = (t & 1) ? hbf_b0 : hbf_b1;
    lstm_step<<<gGemm, 256, 0, stream>>>(aspbf, idxa, wihb2, whhb2, bih_b, bhh_b,
                                         hinB, houB, hbst, cbst, asp_out, alen, t, Ac, 1);
  }

  // position weighting of context
  pos_weight_k<<<dim3(cdivu((long)Bc * Sc * H2c, 256)), 256, 0, stream>>>(ctx_out, clen, llen, alen);

  const dim3 gBH(cdivu((long)Bc * H2c, 256));

  // mca path
  avg_seq_k<<<gBH, 256, 0, stream>>>(asp_out, alen, aavg, Ac);
  matvec_k<<<gBH, 256, 0, stream>>>(aavg, w1, s1v);
  rowdot_k<<<dim3(Bc * Sc), 64, 0, stream>>>(ctx_out, s1v, nullptr, sc1, Sc);
  softmax_rows_k<<<dim3(Bc), 128, 0, stream>>>(sc1, Sc);
  wsum_k<<<gBH, 256, 0, stream>>>(sc1, ctx_out, mca, Sc);

  // mcc path
  avg_seq_k<<<gBH, 256, 0, stream>>>(ctx_out, clen, cavg, Sc);
  matvec_k<<<gBH, 256, 0, stream>>>(cavg, w2, s2v);
  rowdot_k<<<dim3(Bc * Ac), 64, 0, stream>>>(asp_out, s2v, nullptr, sc2, Ac);
  softmax_rows_k<<<dim3(Bc), 128, 0, stream>>>(sc2, Ac);
  wsum_k<<<gBH, 256, 0, stream>>>(sc2, asp_out, mcc, Ac);

  // fine-grained attention tensor u
  rowdot_k<<<dim3(Bc * Sc), 64, 0, stream>>>(ctx_out, nullptr, fc1_w, ucb, Sc);
  rowdot_k<<<dim3(Bc * Ac), 64, 0, stream>>>(asp_out, nullptr, fc1_w + H2c, uab, Ac);
  u_kernel<<<dim3(Bc * Sc), 64, 0, stream>>>(ctx_out, asp_out, ucb, uab, fc1_w + 2 * H2c, fc1_b, ub);

  // mfa
  umax_k<<<dim3(cdivu((long)Bc * Sc, 256)), 256, 0, stream>>>(ub, umx);
  softmax_rows_k<<<dim3(Bc), 128, 0, stream>>>(umx, Sc);
  wsum_k<<<gBH, 256, 0, stream>>>(umx, ctx_out, mfa, Sc);

  // mfc
  softmax8_k<<<dim3(cdivu((long)Bc * Sc, 256)), 256, 0, stream>>>(ub);
  mfc_kernel<<<gBH, 256, 0, stream>>>(ub, asp_out, mfcv);

  // classifier + softmax
  final_fc<<<dim3(1), 256, 0, stream>>>(mca, mcc, mfa, mfcv, fc2_w, fc2_b, outp);
}
